// QLSTM_65481071400784
// MI455X (gfx1250) — matrix operand order repaired and checked by reading, not compiled, not run on hardware
//
#include <hip/hip_runtime.h>
#include <hip/hip_bf16.h>
#include <stdint.h>

#define SEQ    2048
#define BATCH  64
#define IN_DIM 512
#define HID    512
#define CAT    1024
#define NWG    32              // persistent WGs; WG w owns hidden cols [16w,16w+16)
#define TPB    256             // 8 wave32s

typedef __attribute__((ext_vector_type(16))) __bf16 v16bf;
typedef __attribute__((ext_vector_type(8)))  __bf16 v8bf;
typedef __attribute__((ext_vector_type(8)))  float  v8f;

// ---------------------------------------------------------------------------
// ws layout (bytes):
//   [0, 4MB)              : packed bf16 weights, split [w][half][gate][kc][lane][j]
//   [4MB, +64KB)          : H ping (bf16 64x512)
//   [4MB+64KB, +64KB)     : H pong
//   [4MB+128KB, +8)       : grid barrier {count, gen}
//   [8MB, +512MB)         : Zx preactivations, bf16, C-fragment lane layout
// ---------------------------------------------------------------------------
#define WS_WPACK_OFF 0
#define WS_H0_OFF    (4u << 20)
#define WS_H1_OFF    ((4u << 20) + (64u << 10))
#define WS_BAR_OFF   ((4u << 20) + (128u << 10))
#define WS_ZX_OFF    ((size_t)8u << 20)
#define ZX_ELEMS     ((size_t)SEQ * NWG * 4 * 4 * 32 * 8)       // 268M bf16
#define WS_NEED_2PH  (WS_ZX_OFF + ZX_ELEMS * 2)

// Pack W{f,i,g,o} ((CAT,HID) row-major fp32) -> bf16 WMMA-B lane order.
// dest flat id bits: j[4] | lane[5] | kc_h[4] | gate[2] | half[1] | w[5]
//   K = half*512 + kc_h*32 + (lane>=16 ? 8 : 0) + (j<8 ? j : j+8) ; col = w*16 + (lane&15)
__global__ void lstm_pack_weights(const float* __restrict__ Wf,
                                  const float* __restrict__ Wi,
                                  const float* __restrict__ Wg,
                                  const float* __restrict__ Wo,
                                  __bf16* __restrict__ Wp) {
    int id   = blockIdx.x * TPB + threadIdx.x;      // 0 .. 2M-1
    int j    = id & 15;
    int lane = (id >> 4) & 31;
    int kch  = (id >> 9) & 15;
    int g    = (id >> 13) & 3;
    int half = (id >> 15) & 1;
    int w    = id >> 16;
    int col  = w * 16 + (lane & 15);
    int k    = half * 512 + kch * 32 + ((lane & 16) ? 8 : 0) + (j < 8 ? j : j + 8);
    const float* W = (g == 0) ? Wf : (g == 1) ? Wi : (g == 2) ? Wg : Wo;
    Wp[id] = (__bf16)W[k * HID + col];
}

__global__ void lstm_init_state(__bf16* __restrict__ H0,
                                __bf16* __restrict__ H1,
                                unsigned* __restrict__ bar) {
    int id = blockIdx.x * TPB + threadIdx.x;
    if (id < BATCH * HID) { H0[id] = (__bf16)0.0f; H1[id] = (__bf16)0.0f; }
    if (id < 2) bar[id] = 0u;
}

// ------------------------- device helpers ----------------------------------

__device__ __forceinline__ void async_copy_16B(uint32_t lds_addr,
                                               const void* sbase,
                                               uint32_t goff) {
    asm volatile("global_load_async_to_lds_b128 %0, %1, %2"
                 :: "v"(lds_addr), "v"(goff), "s"(sbase) : "memory");
}
__device__ __forceinline__ void wait_async0() {
    asm volatile("s_wait_asynccnt 0" ::: "memory");
}

// A fragment (16x32 bf16) from fp32 x_t in global, converted on the fly.
__device__ __forceinline__ v16bf load_a_x(const float* __restrict__ X,
                                          int t, int mbase, int kc, int lane) {
    int row = mbase + (lane & 15);
    int kb  = kc * 32 + ((lane & 16) ? 8 : 0);
    const float* p = X + ((size_t)t * BATCH + row) * IN_DIM + kb;
    float4 a = *(const float4*)(p);
    float4 b = *(const float4*)(p + 4);
    float4 c = *(const float4*)(p + 16);
    float4 d = *(const float4*)(p + 20);
    v16bf r;
    r[0] = (__bf16)a.x;  r[1] = (__bf16)a.y;  r[2] = (__bf16)a.z;  r[3] = (__bf16)a.w;
    r[4] = (__bf16)b.x;  r[5] = (__bf16)b.y;  r[6] = (__bf16)b.z;  r[7] = (__bf16)b.w;
    r[8] = (__bf16)c.x;  r[9] = (__bf16)c.y;  r[10] = (__bf16)c.z; r[11] = (__bf16)c.w;
    r[12] = (__bf16)d.x; r[13] = (__bf16)d.y; r[14] = (__bf16)d.z; r[15] = (__bf16)d.w;
    return r;
}

// A fragment from a row-major bf16 buffer with row stride `stride` elements.
__device__ __forceinline__ v16bf load_a_bf16(const __bf16* __restrict__ A,
                                             int mbase, int kc, int lane,
                                             int stride) {
    int row = mbase + (lane & 15);
    int kb  = kc * 32 + ((lane & 16) ? 8 : 0);
    const __bf16* p = A + row * stride + kb;
    v8bf lo = *(const v8bf*)(p);
    v8bf hi = *(const v8bf*)(p + 16);
    return __builtin_shufflevector(lo, hi, 0, 1, 2, 3, 4, 5, 6, 7,
                                   8, 9, 10, 11, 12, 13, 14, 15);
}

__device__ __forceinline__ void grid_barrier(unsigned* cnt, unsigned* gen,
                                             unsigned nwg) {
    __builtin_amdgcn_fence(__ATOMIC_RELEASE, "agent");
    __syncthreads();
    if (threadIdx.x == 0) {
        unsigned my  = __hip_atomic_load(gen, __ATOMIC_RELAXED,
                                         __HIP_MEMORY_SCOPE_AGENT);
        unsigned old = __hip_atomic_fetch_add(cnt, 1u, __ATOMIC_ACQ_REL,
                                              __HIP_MEMORY_SCOPE_AGENT);
        if (old == nwg - 1) {
            __hip_atomic_store(cnt, 0u, __ATOMIC_RELAXED,
                               __HIP_MEMORY_SCOPE_AGENT);
            __hip_atomic_fetch_add(gen, 1u, __ATOMIC_RELEASE,
                                   __HIP_MEMORY_SCOPE_AGENT);
        } else {
            while (__hip_atomic_load(gen, __ATOMIC_RELAXED,
                                     __HIP_MEMORY_SCOPE_AGENT) == my) {
                __builtin_amdgcn_s_sleep(2);
            }
        }
    }
    __syncthreads();
    __builtin_amdgcn_fence(__ATOMIC_ACQUIRE, "agent");
}

// ---------------- Phase 1: chip-filling x-preactivation GEMM ----------------
// Block (w, t): all 4 gates x 16 cols x 64 rows for one timestep.
// Zx layout: [t][w][s][mtile 0..3][lane 0..31][r 0..7] bf16 (C-fragment order).
__global__ void __launch_bounds__(TPB, 1)
lstm_xgemm(const float* __restrict__ X,
           const __bf16* __restrict__ Wp,
           __bf16* __restrict__ Zx) {
    __shared__ __align__(16) __bf16 Bl[4 * 16 * 512];   // 64 KB: B x-half slab
    __shared__ __align__(16) __bf16 Ax[64 * 520];       // 66.5 KB: x_t (padded)

    const int w = blockIdx.x, t = blockIdx.y;
    const int tid = threadIdx.x, lane = tid & 31, wv = tid >> 5;
    const int s = wv & 3, mh = wv >> 2;
    const int m0 = mh * 32, m1 = m0 + 16;

    // async DMA: this w's x-half weight slab (contiguous 64KB)
    {
        const char* src = (const char*)Wp + (size_t)w * 131072;
        uint32_t ldsbase = (uint32_t)(uintptr_t)&Bl[0];
        for (int q = tid; q < 4096; q += TPB)
            async_copy_16B(ldsbase + q * 16u, src, (uint32_t)q * 16u);
    }
    // stage x_t as bf16 into LDS (rows padded to 520 to spread banks)
    {
        const float* xt = X + (size_t)t * BATCH * IN_DIM;
        for (int c = tid; c < 4096; c += TPB) {     // 8 floats per chunk
            int row = c >> 6, kk = (c & 63) * 8;
            float4 a = *(const float4*)(xt + row * IN_DIM + kk);
            float4 b = *(const float4*)(xt + row * IN_DIM + kk + 4);
            v8bf z;
            z[0] = (__bf16)a.x; z[1] = (__bf16)a.y; z[2] = (__bf16)a.z; z[3] = (__bf16)a.w;
            z[4] = (__bf16)b.x; z[5] = (__bf16)b.y; z[6] = (__bf16)b.z; z[7] = (__bf16)b.w;
            *(v8bf*)&Ax[row * 520 + kk] = z;
        }
    }
    wait_async0();
    __syncthreads();

    v8f acc0 = {}; v8f acc1 = {};
#pragma unroll 4
    for (int kc = 0; kc < 16; ++kc) {
        v16bf b  = *(const v16bf*)&Bl[(s * 16 + kc) * 512 + lane * 16];
        v16bf a0 = load_a_bf16(Ax, m0, kc, lane, 520);
        v16bf a1 = load_a_bf16(Ax, m1, kc, lane, 520);
        acc0 = __builtin_amdgcn_wmma_f32_16x16x32_bf16(
                   false, a0, false, b, (short)0, acc0, false, false);
        acc1 = __builtin_amdgcn_wmma_f32_16x16x32_bf16(
                   false, a1, false, b, (short)0, acc1, false, false);
    }

    // store both 16x16 tiles in C-fragment lane order (stream, non-temporal)
    size_t base = (((((size_t)t * NWG + w) * 4 + s) * 4) + mh * 2) * 256 + lane * 8;
    v8bf z0, z1;
#pragma unroll
    for (int r = 0; r < 8; ++r) { z0[r] = (__bf16)acc0[r]; z1[r] = (__bf16)acc1[r]; }
    __builtin_nontemporal_store(z0, (v8bf*)(Zx + base));
    __builtin_nontemporal_store(z1, (v8bf*)(Zx + base + 256));
}

// ---------------- Phase 2: persistent recurrent loop (K=512 only) -----------
__global__ void __launch_bounds__(TPB, 1)
lstm_recurrent(const __bf16* __restrict__ Zx,
               const __bf16* __restrict__ Wp,
               const float* __restrict__ bf_, const float* __restrict__ bi_,
               const float* __restrict__ bg_, const float* __restrict__ bo_,
               __bf16* __restrict__ H0, __bf16* __restrict__ H1,
               unsigned* __restrict__ bar,
               float* __restrict__ out) {
    __shared__ __align__(16) __bf16 Bl[4 * 16 * 512];    // 64 KB: B h-half slab
    __shared__ float gbuf[4][BATCH][16];                 // 16 KB
    __shared__ float cbuf[BATCH * 16];                   // 4 KB

    const int w = blockIdx.x;
    const int tid = threadIdx.x, lane = tid & 31, wv = tid >> 5;
    const int s = wv & 3, mh = wv >> 2;
    const int m0 = mh * 32, m1 = m0 + 16;

    {   // async DMA: h-half weight slab (contiguous 64KB at +32768 elems)
        const char* src = (const char*)Wp + (size_t)w * 131072 + 65536;
        uint32_t ldsbase = (uint32_t)(uintptr_t)&Bl[0];
        for (int q = tid; q < 4096; q += TPB)
            async_copy_16B(ldsbase + q * 16u, src, (uint32_t)q * 16u);
        wait_async0();
    }
    for (int e = tid; e < BATCH * 16; e += TPB) cbuf[e] = 0.0f;
    __syncthreads();

    const float* bptr = (s == 0) ? bf_ : (s == 1) ? bi_ : (s == 2) ? bg_ : bo_;
    const float bias  = bptr[w * 16 + (lane & 15)];
    const size_t OUTS = (size_t)SEQ * BATCH * HID;

    for (int t = 0; t < SEQ; ++t) {
        const __bf16* Hcur = (t & 1) ? H1 : H0;
        __bf16*       Hnxt = (t & 1) ? H0 : H1;

        // seed accumulators from phase-1 preactivations (16B/lane, NT loads)
        size_t zb = (((((size_t)t * NWG + w) * 4 + s) * 4) + mh * 2) * 256 + lane * 8;
        v8bf z0 = __builtin_nontemporal_load((const v8bf*)(Zx + zb));
        v8bf z1 = __builtin_nontemporal_load((const v8bf*)(Zx + zb + 256));
        v8f acc0, acc1;
#pragma unroll
        for (int r = 0; r < 8; ++r) { acc0[r] = (float)z0[r]; acc1[r] = (float)z1[r]; }

        // hx @ Wh : K = 512, B from LDS, A from L2-resident bf16 H buffer
#pragma unroll 4
        for (int kc = 0; kc < 16; ++kc) {
            v16bf b  = *(const v16bf*)&Bl[(s * 16 + kc) * 512 + lane * 16];
            v16bf a0 = load_a_bf16(Hcur, m0, kc, lane, HID);
            v16bf a1 = load_a_bf16(Hcur, m1, kc, lane, HID);
            acc0 = __builtin_amdgcn_wmma_f32_16x16x32_bf16(
                       false, a0, false, b, (short)0, acc0, false, false);
            acc1 = __builtin_amdgcn_wmma_f32_16x16x32_bf16(
                       false, a1, false, b, (short)0, acc1, false, false);
        }

        {   // activation + stash gate tiles
            const int col  = lane & 15;
            const int rofs = (lane & 16) ? 8 : 0;
#pragma unroll
            for (int r = 0; r < 8; ++r) {
                float v0 = acc0[r] + bias;
                float v1 = acc1[r] + bias;
                if (s == 2) { v0 = tanhf(v0); v1 = tanhf(v1); }
                else {
                    v0 = 1.0f / (1.0f + __expf(-v0));
                    v1 = 1.0f / (1.0f + __expf(-v1));
                }
                gbuf[s][m0 + rofs + r][col] = v0;
                gbuf[s][m1 + rofs + r][col] = v1;
            }
        }
        __syncthreads();

        for (int e = tid; e < BATCH * 16; e += TPB) {
            int row = e >> 4, col = e & 15;
            float f = gbuf[0][row][col];
            float i = gbuf[1][row][col];
            float g = gbuf[2][row][col];
            float o = gbuf[3][row][col];
            float c = f * cbuf[e] + i * g;
            float h = o * tanhf(c);
            cbuf[e] = c;
            int j = w * 16 + col;
            out[((size_t)t * BATCH + row) * HID + j] = h;
            Hnxt[row * HID + j] = (__bf16)h;
            if (t == SEQ - 1) {
                out[OUTS + (size_t)row * HID + j] = h;
                out[OUTS + (size_t)BATCH * HID + (size_t)row * HID + j] = c;
            }
        }

        grid_barrier(bar, bar + 1, NWG);
    }
}

// ---------------- Fallback: fully fused persistent kernel -------------------
__global__ void __launch_bounds__(TPB, 1)
lstm_persistent(const float* __restrict__ X,
                const __bf16* __restrict__ Wp,
                const float* __restrict__ bf_, const float* __restrict__ bi_,
                const float* __restrict__ bg_, const float* __restrict__ bo_,
                __bf16* __restrict__ H0, __bf16* __restrict__ H1,
                unsigned* __restrict__ bar,
                float* __restrict__ out) {
    __shared__ __align__(16) __bf16 Wl[2 * 4 * 16 * 512];  // 128 KB both halves
    __shared__ float gbuf[4][BATCH][16];
    __shared__ float cbuf[BATCH * 16];

    const int w = blockIdx.x;
    const int tid = threadIdx.x, lane = tid & 31, wv = tid >> 5;
    const int s = wv & 3, mh = wv >> 2;
    const int m0 = mh * 32, m1 = m0 + 16;

    {
        const char* src = (const char*)Wp + (size_t)w * 131072;
        uint32_t ldsbase = (uint32_t)(uintptr_t)&Wl[0];
        for (int q = tid; q < 8192; q += TPB)
            async_copy_16B(ldsbase + q * 16u, src, (uint32_t)q * 16u);
        wait_async0();
    }
    for (int e = tid; e < BATCH * 16; e += TPB) cbuf[e] = 0.0f;
    __syncthreads();

    const float* bptr = (s == 0) ? bf_ : (s == 1) ? bi_ : (s == 2) ? bg_ : bo_;
    const float bias  = bptr[w * 16 + (lane & 15)];
    const size_t OUTS = (size_t)SEQ * BATCH * HID;

    for (int t = 0; t < SEQ; ++t) {
        const __bf16* Hcur = (t & 1) ? H1 : H0;
        __bf16*       Hnxt = (t & 1) ? H0 : H1;

        v8f acc0 = {}; v8f acc1 = {};
        for (int kc = 0; kc < 16; ++kc) {           // x-half
            v16bf b  = *(const v16bf*)&Wl[((0 * 4 + s) * 16 + kc) * 512 + lane * 16];
            v16bf a0 = load_a_x(X, t, m0, kc, lane);
            v16bf a1 = load_a_x(X, t, m1, kc, lane);
            acc0 = __builtin_amdgcn_wmma_f32_16x16x32_bf16(
                       false, a0, false, b, (short)0, acc0, false, false);
            acc1 = __builtin_amdgcn_wmma_f32_16x16x32_bf16(
                       false, a1, false, b, (short)0, acc1, false, false);
        }
        for (int kc = 0; kc < 16; ++kc) {           // h-half
            v16bf b  = *(const v16bf*)&Wl[((1 * 4 + s) * 16 + kc) * 512 + lane * 16];
            v16bf a0 = load_a_bf16(Hcur, m0, kc, lane, HID);
            v16bf a1 = load_a_bf16(Hcur, m1, kc, lane, HID);
            acc0 = __builtin_amdgcn_wmma_f32_16x16x32_bf16(
                       false, a0, false, b, (short)0, acc0, false, false);
            acc1 = __builtin_amdgcn_wmma_f32_16x16x32_bf16(
                       false, a1, false, b, (short)0, acc1, false, false);
        }

        {
            const int col  = lane & 15;
            const int rofs = (lane & 16) ? 8 : 0;
#pragma unroll
            for (int r = 0; r < 8; ++r) {
                float v0 = acc0[r] + bias;
                float v1 = acc1[r] + bias;
                if (s == 2) { v0 = tanhf(v0); v1 = tanhf(v1); }
                else {
                    v0 = 1.0f / (1.0f + __expf(-v0));
                    v1 = 1.0f / (1.0f + __expf(-v1));
                }
                gbuf[s][m0 + rofs + r][col] = v0;
                gbuf[s][m1 + rofs + r][col] = v1;
            }
        }
        __syncthreads();

        for (int e = tid; e < BATCH * 16; e += TPB) {
            int row = e >> 4, col = e & 15;
            float f = gbuf[0][row][col];
            float i = gbuf[1][row][col];
            float g = gbuf[2][row][col];
            float o = gbuf[3][row][col];
            float c = f * cbuf[e] + i * g;
            float h = o * tanhf(c);
            cbuf[e] = c;
            int j = w * 16 + col;
            out[((size_t)t * BATCH + row) * HID + j] = h;
            Hnxt[row * HID + j] = (__bf16)h;
            if (t == SEQ - 1) {
                out[OUTS + (size_t)row * HID + j] = h;
                out[OUTS + (size_t)BATCH * HID + (size_t)row * HID + j] = c;
            }
        }

        grid_barrier(bar, bar + 1, NWG);
    }
}

// ---------------------------------------------------------------------------

extern "C" void kernel_launch(void* const* d_in, const int* in_sizes, int n_in,
                              void* d_out, int out_size, void* d_ws, size_t ws_size,
                              hipStream_t stream) {
    const float* X  = (const float*)d_in[0];
    const float* Wf = (const float*)d_in[1];
    const float* bf = (const float*)d_in[2];
    const float* Wi = (const float*)d_in[3];
    const float* bi = (const float*)d_in[4];
    const float* Wg = (const float*)d_in[5];
    const float* bg = (const float*)d_in[6];
    const float* Wo = (const float*)d_in[7];
    const float* bo = (const float*)d_in[8];
    float* out = (float*)d_out;

    __bf16*   Wp  = (__bf16*)((char*)d_ws + WS_WPACK_OFF);
    __bf16*   H0  = (__bf16*)((char*)d_ws + WS_H0_OFF);
    __bf16*   H1  = (__bf16*)((char*)d_ws + WS_H1_OFF);
    unsigned* bar = (unsigned*)((char*)d_ws + WS_BAR_OFF);
    __bf16*   Zx  = (__bf16*)((char*)d_ws + WS_ZX_OFF);

    lstm_pack_weights<<<(CAT * 4 * HID) / TPB, TPB, 0, stream>>>(Wf, Wi, Wg, Wo, Wp);
    lstm_init_state<<<(BATCH * HID) / TPB, TPB, 0, stream>>>(H0, H1, bar);

    if (ws_size >= WS_NEED_2PH) {
        // Phase 1: parallel x-preactivation GEMM across the whole chip
        lstm_xgemm<<<dim3(NWG, SEQ), TPB, 0, stream>>>(X, Wp, Zx);
        // Phase 2: latency-minimal recurrent chain (K=512 per step)
        lstm_recurrent<<<NWG, TPB, 0, stream>>>(Zx, Wp, bf, bi, bg, bo,
                                                H0, H1, bar, out);
    } else {
        lstm_persistent<<<NWG, TPB, 0, stream>>>(X, Wp, bf, bi, bg, bo,
                                                 H0, H1, bar, out);
    }
}
